// SimpleRNN_14637248545377
// MI455X (gfx1250) — matrix operand order repaired and checked by reading, not compiled, not run on hardware
//
#include <hip/hip_runtime.h>
#include <hip/hip_bf16.h>

typedef __attribute__((ext_vector_type(16))) _Float16 v16h;
typedef __attribute__((ext_vector_type(8)))  float    v8f;

namespace {
constexpr int kT = 1000;   // timesteps
constexpr int kF = 22;     // input features
constexpr int kH = 10;     // hidden units
constexpr int kC = 4;      // classes
constexpr int kB = 1024;   // batch
}

#define WMMA_F16(A, B, C) \
  __builtin_amdgcn_wmma_f32_16x16x32_f16(false, (A), false, (B), (short)0, (C), false, false)

// Single-instruction ReLU: v_med3_num_f32(x, 0, +inf)
static __device__ inline float relu1(float x) {
  return __builtin_amdgcn_fmed3f(x, 0.0f, __builtin_inff());
}

static __device__ inline v16h zero16h() {
  v16h z;
#pragma unroll
  for (int j = 0; j < 16; ++j) z[j] = (_Float16)0.0f;
  return z;
}

// D-row slot map: unit u -> m = u (u<5) or u+3 (5<=u<10)  => m in {0..4} U {8..12}
static __device__ inline int row_unit(int m) {
  return (m < 5) ? m : ((m >= 8 && m < 13) ? (m - 3) : -1);
}
// h1 K-slots: K<5 -> K ; 16<=K<21 -> K-11 ; else invalid
static __device__ inline int col_h1(int k) {
  return (k < 5) ? k : ((k >= 16 && k < 21) ? (k - 11) : -1);
}
// h2 K-slots: 8<=K<13 -> K-8 ; 24<=K<29 -> K-19 ; else invalid
static __device__ inline int col_h2(int k) {
  return (k >= 8 && k < 13) ? (k - 8) : ((k >= 24 && k < 29) ? (k - 19) : -1);
}

// A fragment 16x32 f16: lane L holds row m=L&15; elem j -> K=j+16*(L>>4)

// W_ih0 (10 x 22): K indexes raw feature.
static __device__ inline v16h make_a_ih0(const float* __restrict__ W, int lane) {
  const int m = lane & 15, half = lane >> 4;
  const int u = row_unit(m);
  v16h a;
#pragma unroll
  for (int j = 0; j < 16; ++j) {
    const int k = j + 16 * half;
    float v = 0.0f;
    if (u >= 0 && k < kF) v = W[u * kF + k];
    a[j] = (_Float16)v;
  }
  return a;
}

// W_hh0 (10 x 10): columns live in the h1 slots only.
static __device__ inline v16h make_a_hh0(const float* __restrict__ W, int lane) {
  const int m = lane & 15, half = lane >> 4;
  const int u = row_unit(m);
  v16h a;
#pragma unroll
  for (int j = 0; j < 16; ++j) {
    const int k = j + 16 * half;
    const int i = col_h1(k);
    float v = 0.0f;
    if (u >= 0 && i >= 0) v = W[u * kH + i];
    a[j] = (_Float16)v;
  }
  return a;
}

// Merged layer-1 A: W_ih1 columns in h1 slots, W_hh1 columns in h2 slots.
static __device__ inline v16h make_a_l1(const float* __restrict__ Wih,
                                        const float* __restrict__ Whh, int lane) {
  const int m = lane & 15, half = lane >> 4;
  const int u = row_unit(m);
  v16h a;
#pragma unroll
  for (int j = 0; j < 16; ++j) {
    const int k = j + 16 * half;
    const int i1 = col_h1(k);
    const int i2 = col_h2(k);
    float v = 0.0f;
    if (u >= 0 && i1 >= 0) v = Wih[u * kH + i1];
    if (u >= 0 && i2 >= 0) v = Whh[u * kH + i2];
    a[j] = (_Float16)v;
  }
  return a;
}

// W_lin (4 x 10): rows = classes at m=0..3; columns read the h2 slots of s.
static __device__ inline v16h make_a_lin(const float* __restrict__ W, int lane) {
  const int m = lane & 15, half = lane >> 4;
  const int c = (m < kC) ? m : -1;
  v16h a;
#pragma unroll
  for (int j = 0; j < 16; ++j) {
    const int k = j + 16 * half;
    const int i = col_h2(k);
    float v = 0.0f;
    if (c >= 0 && i >= 0) v = W[c * kH + i];
    a[j] = (_Float16)v;
  }
  return a;
}

// Per-lane bias fragment for the hidden D layout (m = v + 8*half).
static __device__ inline v8f make_bias_h(const float* __restrict__ b1,
                                         const float* __restrict__ b2, int lane) {
  const int half = lane >> 4;
  v8f c;
#pragma unroll
  for (int v = 0; v < 8; ++v) {
    const int u = row_unit(v + 8 * half);
    c[v] = (u >= 0) ? (b1[u] + b2[u]) : 0.0f;
  }
  return c;
}

static __device__ inline v8f make_bias_lin(const float* __restrict__ b, int lane) {
  const int half = lane >> 4;
  v8f c;
#pragma unroll
  for (int v = 0; v < 8; ++v) {
    const int m = v + 8 * half;
    c[v] = (m < kC) ? b[m] : 0.0f;
  }
  return c;
}

// B fragment (32x16 f16) of x_t^T: lane n (half 0) carries K=0..15,
// lane n+16 (half 1) carries K=16..31; elem j -> K = j + 16*half.
static __device__ inline v16h load_x_frag(const float* __restrict__ xr, int half) {
  const float2* p2 = reinterpret_cast<const float2*>(xr);  // row base is 8B aligned
  v16h b = zero16h();
  if (half == 0) {
#pragma unroll
    for (int q = 0; q < 8; ++q) {            // features 0..15
      const float2 v = p2[q];
      b[2 * q]     = (_Float16)v.x;
      b[2 * q + 1] = (_Float16)v.y;
    }
  } else {
#pragma unroll
    for (int q = 0; q < 3; ++q) {            // features 16..21 -> K=16..21
      const float2 v = p2[8 + q];
      b[2 * q]     = (_Float16)v.x;
      b[2 * q + 1] = (_Float16)v.y;
    }
  }
  return b;
}

__global__ __launch_bounds__(32) void rnn2_wmma_kernel(
    const float* __restrict__ x,
    const float* __restrict__ w_ih0, const float* __restrict__ w_hh0,
    const float* __restrict__ b_ih0, const float* __restrict__ b_hh0,
    const float* __restrict__ w_ih1, const float* __restrict__ w_hh1,
    const float* __restrict__ b_ih1, const float* __restrict__ b_hh1,
    const float* __restrict__ w_lin, const float* __restrict__ b_lin,
    float* __restrict__ out) {
  const int lane = threadIdx.x & 31;
  const int half = lane >> 4;
  // Two independent 16-row batch tiles per wave to hide WMMA hazard latency.
  const int rowA = blockIdx.x * 32 + (lane & 15);
  const int rowB = rowA + 16;

  const v16h a_ih0 = make_a_ih0(w_ih0, lane);
  const v16h a_hh0 = make_a_hh0(w_hh0, lane);
  const v16h a_l1  = make_a_l1(w_ih1, w_hh1, lane);
  const v16h a_lin = make_a_lin(w_lin, lane);
  const v8f bias0 = make_bias_h(b_ih0, b_hh0, lane);
  const v8f bias1 = make_bias_h(b_ih1, b_hh1, lane);
  const v8f biasl = make_bias_lin(b_lin, lane);

  const float* xbaseA = x + (size_t)rowA * (kT * kF);
  const float* xbaseB = x + (size_t)rowB * (kT * kF);

  // Persistent state fragment per chain: h1 in elems 0..4 (K {0..4,16..20}),
  // h2 in elems 8..12 (K {8..12,24..28}), everything else stays zero.
  v16h sA = zero16h(), sB = zero16h();
  v16h bxA = load_x_frag(xbaseA, half);
  v16h bxB = load_x_frag(xbaseB, half);

  const int cap = (kT - 1) * kF;     // clamped element offsets (strength-reduced)
  int xoff = 0;

  for (int t = 0; t < kT; ++t) {
    const int xn = (xoff + kF < cap) ? (xoff + kF) : cap;
    const int xp = (xoff + 8 * kF < cap) ? (xoff + 8 * kF) : cap;
    // Software pipeline: next-step x fragments + prefetch overlap the WMMA chain.
    v16h nxA = load_x_frag(xbaseA + xn, half);
    v16h nxB = load_x_frag(xbaseB + xn, half);
    __builtin_prefetch(xbaseA + xp, 0, 1);
    __builtin_prefetch(xbaseB + xp, 0, 1);

    // Layer 0 (2 WMMAs: feature term + recurrent term), chains interleaved.
    v8f c1A = WMMA_F16(a_ih0, bxA, bias0);
    v8f c1B = WMMA_F16(a_ih0, bxB, bias0);
    c1A = WMMA_F16(a_hh0, sA, c1A);
    c1B = WMMA_F16(a_hh0, sB, c1B);
#pragma unroll
    for (int j = 0; j < 5; ++j) {       // h1_t -> elems 0..4 (in place)
      sA[j] = (_Float16)relu1(c1A[j]);
      sB[j] = (_Float16)relu1(c1B[j]);
    }

    // Layer 1 merged into ONE WMMA: A = [W_ih1 | W_hh1], B = s (h1_t, h2_{t-1}).
    v8f c2A = WMMA_F16(a_l1, sA, bias1);
    v8f c2B = WMMA_F16(a_l1, sB, bias1);
#pragma unroll
    for (int j = 0; j < 5; ++j) {       // h2_t -> elems 8..12 (in place)
      sA[8 + j] = (_Float16)relu1(c2A[j]);
      sB[8 + j] = (_Float16)relu1(c2B[j]);
    }

    bxA = nxA;
    bxB = nxB;
    xoff = xn;
  }

  // Head: out^T (4 x 16) = W_lin @ h2^T ; h2 already >= 0 so ReLU is idempotent.
  v8f oA = WMMA_F16(a_lin, sA, biasl);
  v8f oB = WMMA_F16(a_lin, sB, biasl);

  if (lane < 16) {
    const float4 stA = make_float4(oA[0], oA[1], oA[2], oA[3]);
    const float4 stB = make_float4(oB[0], oB[1], oB[2], oB[3]);
    *reinterpret_cast<float4*>(out + (size_t)rowA * kC) = stA;
    *reinterpret_cast<float4*>(out + (size_t)rowB * kC) = stB;
  }
}

extern "C" void kernel_launch(void* const* d_in, const int* in_sizes, int n_in,
                              void* d_out, int out_size, void* d_ws, size_t ws_size,
                              hipStream_t stream) {
  (void)in_sizes; (void)n_in; (void)out_size; (void)d_ws; (void)ws_size;
  const float* x     = (const float*)d_in[0];
  const float* w_ih0 = (const float*)d_in[1];
  const float* w_hh0 = (const float*)d_in[2];
  const float* b_ih0 = (const float*)d_in[3];
  const float* b_hh0 = (const float*)d_in[4];
  const float* w_ih1 = (const float*)d_in[5];
  const float* w_hh1 = (const float*)d_in[6];
  const float* b_ih1 = (const float*)d_in[7];
  const float* b_hh1 = (const float*)d_in[8];
  const float* w_lin = (const float*)d_in[9];
  const float* b_lin = (const float*)d_in[10];
  float* out = (float*)d_out;

  // 1024 batch rows / 32 rows per wave = 32 single-wave workgroups.
  rnn2_wmma_kernel<<<dim3(kB / 32), dim3(32), 0, stream>>>(
      x, w_ih0, w_hh0, b_ih0, b_hh0, w_ih1, w_hh1, b_ih1, b_hh1, w_lin, b_lin, out);
}
